// MultiScaleAttention_30958124270272
// MI455X (gfx1250) — compile-verified, matrix operand order checked by reading, not run on hardware
//
#include <hip/hip_runtime.h>
#include <hip/hip_bf16.h>
#include <math.h>

typedef __bf16 bf16_t;
typedef __attribute__((ext_vector_type(16))) __bf16 v16bf;
typedef __attribute__((ext_vector_type(8)))  __bf16 v8bf;
typedef __attribute__((ext_vector_type(8)))  float  v8f;
typedef __attribute__((ext_vector_type(4)))  int    v4i;

#define DIM       768
#define HEAD_DIM  64
#define LOCAL_HEAD 4
#define LOCAL_C   256
#define WS        7
#define NR        49
#define BATCH     32
#define HIMG      56
#define WIMG      56
#define NTOK      (HIMG*WIMG)     // 3136
#define MROWS     (BATCH*NTOK)    // 100352
#define QKV_N     (3*DIM)         // 2304

#if __has_builtin(__builtin_amdgcn_global_load_async_to_lds_b128)
#define HAS_ASYNC_LDS 1
#else
#define HAS_ASYNC_LDS 0
#endif

typedef __attribute__((address_space(1))) v4i* gv4i_p;
typedef __attribute__((address_space(3))) v4i* lv4i_p;

static __device__ __forceinline__ v16bf frag_combine(v8bf lo, v8bf hi) {
  return __builtin_shufflevector(lo, hi, 0,1,2,3,4,5,6,7,8,9,10,11,12,13,14,15);
}

// 16-byte global -> LDS copy: async DMA on CDNA5, sync fallback otherwise.
static __device__ __forceinline__ void cp16_g2l(const bf16_t* g, bf16_t* l) {
#if HAS_ASYNC_LDS
  __builtin_amdgcn_global_load_async_to_lds_b128((gv4i_p)g, (lv4i_p)l, 0, 0);
#else
  *(v8bf*)l = *(const v8bf*)g;
#endif
}

template<int N>
static __device__ __forceinline__ void wait_async() {
#if HAS_ASYNC_LDS
#if __has_builtin(__builtin_amdgcn_s_wait_asynccnt)
  __builtin_amdgcn_s_wait_asynccnt((unsigned short)N);
#else
  asm volatile("s_wait_asynccnt %0" :: "i"(N) : "memory");
#endif
#endif
}

// ---------------------------------------------------------------- convert
__global__ void cvt_f32_bf16(const float* __restrict__ src, bf16_t* __restrict__ dst, long n) {
  long i = (long)blockIdx.x * blockDim.x + threadIdx.x;
  if (i < n) dst[i] = (bf16_t)src[i];
}

// ---------------------------------------------------------------- big GEMM
// C[M,N] = A[M,K] * Bw[N,K]^T   (both K-contiguous bf16), f32 accumulate.
// Workgroup: 256 thr = 8 waves (2 M x 4 N), wave tile 64x64 (4x4 WMMA frags).
// K is staged in 32-wide double-buffered LDS tiles via async global->LDS DMA.
#define APAD 40   // LDS row stride (elements): 80B = 20 banks -> conflict-free b128 reads
#define A_TILE (128*APAD)
#define B_TILE (256*APAD)

template<bool STORE_BF16>
__global__ void __launch_bounds__(256) gemm_wmma_bf16(
    const bf16_t* __restrict__ A, const bf16_t* __restrict__ Bw,
    bf16_t* __restrict__ outB, float* __restrict__ outF,
    const float* __restrict__ bias, int M, int N, int K)
{
  __shared__ bf16_t As[2*A_TILE];
  __shared__ bf16_t Bs[2*B_TILE];

  const int tid  = threadIdx.x;
  const int lane = tid & 31;
  const int wave = tid >> 5;
  const int wm = wave >> 2, wn = wave & 3;
  const int lr = lane & 15, lg = lane >> 4;
  const long m_base = (long)blockIdx.y * 128;
  const long n_base = (long)blockIdx.x * 256;

  // stage one 32-wide K-slice of A(128 rows) and B(256 rows) into LDS buffer `buf`
  auto prefetch = [&](int buf, int k0) {
#pragma unroll
    for (int t = 0; t < 2; ++t) {           // A: 512 x 16B chunks
      int c = tid + t*256;
      int row = c >> 2, cc = (c & 3) * 8;
      cp16_g2l(A + (m_base + row)*(long)K + k0 + cc,
               As + buf*A_TILE + row*APAD + cc);
    }
#pragma unroll
    for (int t = 0; t < 4; ++t) {           // B: 1024 x 16B chunks
      int c = tid + t*256;
      int row = c >> 2, cc = (c & 3) * 8;
      cp16_g2l(Bw + (n_base + row)*(long)K + k0 + cc,
               Bs + buf*B_TILE + row*APAD + cc);
    }
  };

  v8f acc[4][4] = {};
  const int nk = K / 32;
  prefetch(0, 0);
  for (int ks = 0; ks < nk; ++ks) {
    const int buf = ks & 1;
    if (ks + 1 < nk) { prefetch(buf ^ 1, (ks + 1) * 32); wait_async<6>(); }
    else             { wait_async<0>(); }
    __syncthreads();   // all waves' current-buffer tiles visible

    const bf16_t* Ab = As + buf*A_TILE + (wm*64)*APAD;
    const bf16_t* Bb = Bs + buf*B_TILE + (wn*64)*APAD;
    v16bf af[4];
#pragma unroll
    for (int i = 0; i < 4; ++i) {
      const bf16_t* p = Ab + (i*16 + lr)*APAD + lg*8;
      af[i] = frag_combine(*(const v8bf*)p, *(const v8bf*)(p + 16));
    }
#pragma unroll
    for (int j = 0; j < 4; ++j) {
      const bf16_t* p = Bb + (j*16 + lr)*APAD + lg*8;
      v16bf bfr = frag_combine(*(const v8bf*)p, *(const v8bf*)(p + 16));
#pragma unroll
      for (int i = 0; i < 4; ++i)
        acc[i][j] = __builtin_amdgcn_wmma_f32_16x16x32_bf16(
            false, af[i], false, bfr, (short)0, acc[i][j], false, false);
    }
    __syncthreads();   // done reading before next prefetch overwrites this buffer
  }

#pragma unroll
  for (int i = 0; i < 4; ++i)
#pragma unroll
    for (int j = 0; j < 4; ++j)
#pragma unroll
      for (int e = 0; e < 8; ++e) {
        long row = m_base + wm*64 + i*16 + e + 8*lg;
        long col = n_base + wn*64 + j*16 + lr;
        if (STORE_BF16) outB[row * (long)N + col] = (bf16_t)acc[i][j][e];
        else            outF[row * (long)N + col] = acc[i][j][e] + bias[col];
      }
}

// ---------------------------------------------------------------- pool (avg+max) + pack for scales 1,2
// Output: per (j,b,region,head) a 64x64 bf16 tile (rows 49..63 zero).
__global__ void pool_pack(const bf16_t* __restrict__ qkvb, bf16_t* __restrict__ packed,
                          int scaleIdx, int nreg, int Wi)
{
  long id = (long)blockIdx.x * blockDim.x + threadIdx.x;
  long total = 3L * BATCH * nreg * LOCAL_HEAD * 64 * 64;
  if (id >= total) return;
  int d = (int)(id & 63);
  int p = (int)((id >> 6) & 63);
  long t2 = id >> 12;
  int h = (int)(t2 & 3); t2 >>= 2;
  int r = (int)(t2 % nreg); t2 /= nreg;
  int b = (int)(t2 % BATCH);
  int j = (int)(t2 / BATCH);
  long tile = (((long)j*BATCH + b)*nreg + r)*LOCAL_HEAD + h;

  bf16_t out = (bf16_t)0.f;
  if (p < NR) {
    int s  = r*NR + p;
    int yy = s / Wi, xx = s % Wi;
    int kk = 1 << scaleIdx;
    float sum = 0.f, mx = -3.4e38f;
    for (int dy = 0; dy < kk; ++dy)
      for (int dx = 0; dx < kk; ++dx) {
        int tok = (yy*kk + dy)*WIMG + (xx*kk + dx);
        float v = (float)qkvb[((long)b*NTOK + tok)*QKV_N + j*DIM + scaleIdx*LOCAL_C + h*HEAD_DIM + d];
        sum += v; mx = fmaxf(mx, v);
      }
    out = (bf16_t)(sum / (float)(kk*kk) + mx);
  }
  packed[tile*4096 + p*64 + d] = out;
}

// ---------------------------------------------------------------- window attention
// One workgroup (128 thr = 4 waves) per (b, region, head).
// mode 0: read q/k/v straight from qkv buffer (scale 0), write fused bf16.
// mode 1: read packed pooled tiles, write y fp32 [b][ch][region*49+p].
__global__ void __launch_bounds__(128) attn_win(
    const bf16_t* __restrict__ qkvb,
    const bf16_t* __restrict__ packed,
    const float*  __restrict__ rpb,
    bf16_t* __restrict__ fusedOut,
    float*  __restrict__ yOut,
    int nreg, int mode)
{
  __shared__ bf16_t qs[64*64];
  __shared__ bf16_t ks[64*64];
  __shared__ bf16_t vts[64*64];   // v transposed: vts[d][m]
  __shared__ float  Sf[64][65];
  __shared__ bf16_t Ps[64*64];

  int id = blockIdx.x;
  int h  = id & 3;
  int r  = (id >> 2) % nreg;
  int b  = id / (4*nreg);
  int tid = threadIdx.x;

  // stage q, k, v^T into LDS (rows >= 49 zero)
  for (int c = tid; c < 512; c += 128) {
    int row = c >> 3;
    int cc  = (c & 7) << 3;
    v8bf qv = {}, kv = {}, vv = {};
    if (mode == 0) {
      if (row < NR) {
        long base = ((long)b*NTOK + (long)r*NR + row)*QKV_N + h*HEAD_DIM + cc;
        qv = *(const v8bf*)(qkvb + base);
        kv = *(const v8bf*)(qkvb + base + DIM);
        vv = *(const v8bf*)(qkvb + base + 2*DIM);
      }
    } else {
      long tstride = (long)BATCH*nreg*LOCAL_HEAD*4096;
      long base = (((long)b*nreg + r)*LOCAL_HEAD + h)*4096 + row*64 + cc;
      qv = *(const v8bf*)(packed + base);
      kv = *(const v8bf*)(packed + base + tstride);
      vv = *(const v8bf*)(packed + base + 2*tstride);
    }
    *(v8bf*)(qs + row*64 + cc) = qv;
    *(v8bf*)(ks + row*64 + cc) = kv;
#pragma unroll
    for (int e = 0; e < 8; ++e) vts[(cc+e)*64 + row] = vv[e];
  }
  __syncthreads();

  int lane = tid & 31, w = tid >> 5;
  int lr = lane & 15, lg = lane >> 4;

  // S = q * k^T  (64x64 padded, wave w owns rows 16w..16w+15)
  {
    v8f sacc[4] = {};
#pragma unroll
    for (int kt = 0; kt < 2; ++kt) {
      int k0 = kt*32 + lg*8;
      const bf16_t* ap = qs + (w*16 + lr)*64 + k0;
      v16bf aq = frag_combine(*(const v8bf*)ap, *(const v8bf*)(ap+16));
#pragma unroll
      for (int nt = 0; nt < 4; ++nt) {
        const bf16_t* bp = ks + (nt*16 + lr)*64 + k0;
        v16bf bk = frag_combine(*(const v8bf*)bp, *(const v8bf*)(bp+16));
        sacc[nt] = __builtin_amdgcn_wmma_f32_16x16x32_bf16(
            false, aq, false, bk, (short)0, sacc[nt], false, false);
      }
    }
#pragma unroll
    for (int nt = 0; nt < 4; ++nt)
#pragma unroll
      for (int e = 0; e < 8; ++e)
        Sf[w*16 + e + 8*lg][nt*16 + lr] = sacc[nt][e];
  }
  __syncthreads();

  // softmax with relative-position bias (scale = 64^-0.5 = 0.125)
  if (tid < 64) {
    int n = tid;
    if (n < NR) {
      int rn = n / WS, cn = n % WS;
      float mx = -3.4e38f;
      for (int m = 0; m < NR; ++m) {
        int rm = m / WS, cm = m % WS;
        int idx = (rn - rm + WS - 1)*(2*WS - 1) + (cn - cm + WS - 1);
        float v = Sf[n][m]*0.125f + rpb[idx*LOCAL_HEAD + h];
        Sf[n][m] = v;
        mx = fmaxf(mx, v);
      }
      float s = 0.f;
      for (int m = 0; m < NR; ++m) { float e = __expf(Sf[n][m] - mx); s += e; Sf[n][m] = e; }
      float inv = 1.f / s;
      for (int m = 0;  m < NR; ++m) Ps[n*64 + m] = (bf16_t)(Sf[n][m]*inv);
      for (int m = NR; m < 64; ++m) Ps[n*64 + m] = (bf16_t)0.f;
    } else {
      for (int m = 0; m < 64; ++m) Ps[n*64 + m] = (bf16_t)0.f;
    }
  }
  __syncthreads();

  // y = P * v   (B fragments come from pre-transposed v)
  v8f yacc[4] = {};
#pragma unroll
  for (int kt = 0; kt < 2; ++kt) {
    int k0 = kt*32 + lg*8;
    const bf16_t* ap = Ps + (w*16 + lr)*64 + k0;
    v16bf af = frag_combine(*(const v8bf*)ap, *(const v8bf*)(ap+16));
#pragma unroll
    for (int nt = 0; nt < 4; ++nt) {
      const bf16_t* bp = vts + (nt*16 + lr)*64 + k0;
      v16bf bv = frag_combine(*(const v8bf*)bp, *(const v8bf*)(bp+16));
      yacc[nt] = __builtin_amdgcn_wmma_f32_16x16x32_bf16(
          false, af, false, bv, (short)0, yacc[nt], false, false);
    }
  }

  int npix = nreg * NR;
#pragma unroll
  for (int nt = 0; nt < 4; ++nt)
#pragma unroll
    for (int e = 0; e < 8; ++e) {
      int p = w*16 + e + 8*lg;
      if (p < NR) {
        int d = nt*16 + lr;
        if (mode == 0)
          fusedOut[((long)b*NTOK + (long)r*NR + p)*DIM + h*HEAD_DIM + d] = (bf16_t)yacc[nt][e];
        else
          yOut[((long)b*LOCAL_C + h*HEAD_DIM + d)*(long)npix + r*NR + p] = yacc[nt][e];
      }
    }
}

// ---------------------------------------------------------------- bilinear upsample into fused
__global__ void upsample_bilinear(const float* __restrict__ yIn, bf16_t* __restrict__ fused,
                                  int scaleIdx, int Hi)
{
  long id = (long)blockIdx.x * blockDim.x + threadIdx.x;
  if (id >= (long)BATCH*LOCAL_C*NTOK) return;
  int t  = (int)(id % NTOK);
  int ch = (int)((id / NTOK) % LOCAL_C);
  int b  = (int)(id / ((long)NTOK*LOCAL_C));
  int oy = t / WIMG, ox = t % WIMG;
  float sc = (float)Hi / (float)HIMG;
  float fy = ((float)oy + 0.5f)*sc - 0.5f;
  float fx = ((float)ox + 0.5f)*sc - 0.5f;
  int y0 = (int)floorf(fy), x0 = (int)floorf(fx);
  float wy = fy - (float)y0, wx = fx - (float)x0;
  int y0c = min(max(y0, 0), Hi-1), y1c = min(max(y0+1, 0), Hi-1);
  int x0c = min(max(x0, 0), Hi-1), x1c = min(max(x0+1, 0), Hi-1);
  const float* base = yIn + ((long)b*LOCAL_C + ch)*(long)(Hi*Hi);
  float v00 = base[y0c*Hi + x0c], v01 = base[y0c*Hi + x1c];
  float v10 = base[y1c*Hi + x0c], v11 = base[y1c*Hi + x1c];
  float v = (1.f-wy)*((1.f-wx)*v00 + wx*v01) + wy*((1.f-wx)*v10 + wx*v11);
  fused[((long)b*NTOK + t)*DIM + scaleIdx*LOCAL_C + ch] = (bf16_t)v;
}

// ---------------------------------------------------------------- launch
extern "C" void kernel_launch(void* const* d_in, const int* in_sizes, int n_in,
                              void* d_out, int out_size, void* d_ws, size_t ws_size,
                              hipStream_t stream)
{
  (void)in_sizes; (void)n_in; (void)out_size; (void)ws_size;
  const float* x      = (const float*)d_in[0];
  const float* qkv_w  = (const float*)d_in[1];
  const float* proj_w = (const float*)d_in[2];
  const float* proj_b = (const float*)d_in[3];
  const float* rpb    = (const float*)d_in[4];

  char* ws = (char*)d_ws;
  size_t off = 0;
  auto alloc = [&](size_t bytes) -> void* {
    void* p = ws + off;
    off = (off + bytes + 255) & ~(size_t)255;
    return p;
  };
  bf16_t* xb     = (bf16_t*)alloc((size_t)MROWS*DIM*2);
  bf16_t* qkvwb  = (bf16_t*)alloc((size_t)QKV_N*DIM*2);
  bf16_t* projwb = (bf16_t*)alloc((size_t)DIM*DIM*2);
  bf16_t* qkvb   = (bf16_t*)alloc((size_t)MROWS*QKV_N*2);
  bf16_t* pk1    = (bf16_t*)alloc((size_t)3*BATCH*16*LOCAL_HEAD*4096*2);
  bf16_t* pk2    = (bf16_t*)alloc((size_t)3*BATCH*4 *LOCAL_HEAD*4096*2);
  float*  y1     = (float*) alloc((size_t)BATCH*LOCAL_C*784*4);
  float*  y2     = (float*) alloc((size_t)BATCH*LOCAL_C*196*4);
  bf16_t* fused  = (bf16_t*)alloc((size_t)MROWS*DIM*2);

  { long n = (long)MROWS*DIM;
    cvt_f32_bf16<<<(unsigned)((n + 255)/256), 256, 0, stream>>>(x, xb, n); }
  { long n = (long)QKV_N*DIM;
    cvt_f32_bf16<<<(unsigned)((n + 255)/256), 256, 0, stream>>>(qkv_w, qkvwb, n); }
  { long n = (long)DIM*DIM;
    cvt_f32_bf16<<<(unsigned)((n + 255)/256), 256, 0, stream>>>(proj_w, projwb, n); }

  // QKV projection: 100352 x 2304 x 768
  gemm_wmma_bf16<true><<<dim3(QKV_N/256, MROWS/128), 256, 0, stream>>>(
      xb, qkvwb, qkvb, nullptr, nullptr, MROWS, QKV_N, DIM);

  // pooled + packed q/k/v for scales 1 and 2
  { long tot = 3L*BATCH*16*LOCAL_HEAD*4096;
    pool_pack<<<(unsigned)((tot+255)/256), 256, 0, stream>>>(qkvb, pk1, 1, 16, 28); }
  { long tot = 3L*BATCH*4*LOCAL_HEAD*4096;
    pool_pack<<<(unsigned)((tot+255)/256), 256, 0, stream>>>(qkvb, pk2, 2, 4, 14); }

  // window attention per scale
  attn_win<<<BATCH*64*LOCAL_HEAD, 128, 0, stream>>>(qkvb, nullptr, rpb, fused, nullptr, 64, 0);
  attn_win<<<BATCH*16*LOCAL_HEAD, 128, 0, stream>>>(nullptr, pk1,  rpb, nullptr, y1, 16, 1);
  attn_win<<<BATCH*4 *LOCAL_HEAD, 128, 0, stream>>>(nullptr, pk2,  rpb, nullptr, y2, 4,  1);

  // upsample scales 1,2 into fused
  { long tot = (long)BATCH*LOCAL_C*NTOK;
    upsample_bilinear<<<(unsigned)((tot+255)/256), 256, 0, stream>>>(y1, fused, 1, 28);
    upsample_bilinear<<<(unsigned)((tot+255)/256), 256, 0, stream>>>(y2, fused, 2, 14); }

  // output projection: 100352 x 768 x 768 (+bias) -> fp32 d_out
  gemm_wmma_bf16<false><<<dim3(DIM/256, MROWS/128), 256, 0, stream>>>(
      fused, projwb, nullptr, (float*)d_out, proj_b, MROWS, DIM, DIM);
}
